// EnhancedConformationalConsistencyLoss_15607911153862
// MI455X (gfx1250) — matrix operand order repaired and checked by reading, not compiled, not run on hardware
//
#include <hip/hip_runtime.h>
#include <hip/hip_bf16.h>
#include <math.h>

// ---------------------------------------------------------------------------
// CDNA5 (gfx1250) implementation: bf16 WMMA GEMMs + fragment-sorted Gram tiles
// ---------------------------------------------------------------------------

typedef __attribute__((ext_vector_type(16))) __bf16 v16bf;
typedef __attribute__((ext_vector_type(8)))  float  v8f;

#define HDIM   256
#define NFRAG  256
#define LOG2_F 0.69314718055994530942f
#define R_MIX  0.3f
#define C_CONS 0.8f

__device__ __forceinline__ unsigned short f2bf(float f) {
  unsigned u = __float_as_uint(f);
  u += 0x7FFFu + ((u >> 16) & 1u);          // round-to-nearest-even
  return (unsigned short)(u >> 16);
}

union FragU { v16bf v; uint4 q[2]; };

// A-matrix 16x32 bf16 fragment (ISA layout): lane = 16*(khalf)+m.
// elements 0..7  <-> k = kb..kb+7      (kb = 8 for lanes 16..31)
// elements 8..15 <-> k = 16+kb..16+kb+7
__device__ __forceinline__ v16bf load_a_frag(const unsigned short* __restrict__ A,
                                             int row, int k0, int lane) {
  const unsigned short* p = A + (size_t)row * HDIM + k0 + ((lane >> 4) << 3);
  FragU f;
  f.q[0] = *reinterpret_cast<const uint4*>(p);
  f.q[1] = *reinterpret_cast<const uint4*>(p + 16);
  return f.v;
}

// B-matrix 32x16 bf16 fragment from a TRANSPOSED operand Bt[n][k]:
// lane = 16*(khalf)+n, elements 0..7 <-> k = kb..kb+7, 8..15 <-> k = 16+kb..16+kb+7  (kb = 8 for lanes 16..31)
__device__ __forceinline__ v16bf load_b_frag(const unsigned short* __restrict__ Bt,
                                             int row, int k0, int lane) {
  const unsigned short* p = Bt + (size_t)row * HDIM + k0 + ((lane >> 4) << 3);
  FragU f;
  f.q[0] = *reinterpret_cast<const uint4*>(p);
  f.q[1] = *reinterpret_cast<const uint4*>(p + 16);
  return f.v;
}

// -------------------------------- prep kernels -----------------------------

__global__ void mix_bf16_kernel(const float* __restrict__ a, const float* __restrict__ b,
                                unsigned short* __restrict__ o, int count) {
  for (int i = blockIdx.x * blockDim.x + threadIdx.x; i < count;
       i += gridDim.x * blockDim.x)
    o[i] = f2bf(R_MIX * a[i] + (1.0f - R_MIX) * b[i]);
}

// W[k][n] (row-major HxH f32) -> Wt[n][k] bf16
__global__ void transpose_bf16_kernel(const float* __restrict__ W,
                                      unsigned short* __restrict__ Wt) {
  int idx = blockIdx.x * 256 + threadIdx.x;      // 65536 total
  int n = idx >> 8, k = idx & 255;
  Wt[idx] = f2bf(W[k * HDIM + n]);
}

// zero the 64 padded rows of each of the 4 z-sets
__global__ void zero_tail_kernel(unsigned short* __restrict__ Z, int zStride, int n) {
  int idx = blockIdx.x * 256 + threadIdx.x;      // 4*64*256 = 65536 total
  int s = idx / (64 * 256);
  int rem = idx - s * (64 * 256);
  Z[(size_t)s * zStride + (size_t)n * HDIM + rem] = 0;
}

// ------------------------- counting sort (deterministic) -------------------

__global__ void hist_scan_kernel(const int* __restrict__ frag,
                                 int* __restrict__ counts, int* __restrict__ starts, int n) {
  __shared__ int lc[NFRAG];
  int t = threadIdx.x;
  lc[t] = 0;
  __syncthreads();
  for (int i = t; i < n; i += 256) atomicAdd(&lc[frag[i] & (NFRAG - 1)], 1);
  __syncthreads();
  counts[t] = lc[t];
  if (t == 0) {
    int acc = 0;
    for (int f = 0; f < NFRAG; ++f) { starts[f] = acc; acc += lc[f]; }
  }
}

// stable rank within fragment -> deterministic sorted order
__global__ void rank_scatter_kernel(const int* __restrict__ frag,
                                    const int* __restrict__ starts,
                                    int* __restrict__ order, int n) {
  int i = blockIdx.x * blockDim.x + threadIdx.x;
  if (i >= n) return;
  int f = frag[i];
  int r = 0;
  for (int j = 0; j < i; ++j) r += (frag[j] == f) ? 1 : 0;
  order[starts[f & (NFRAG - 1)] + r] = i;
}

// ------------------------------ WMMA GEMM ----------------------------------
// C[M,256] = A[M,256] * B (B given transposed: Bt[n][k]), + bias
// Each wave computes a 16x64 output strip (4 accumulators): the A fragment is
// loaded once per K-chunk and reused by 4 WMMAs (2.5 b128 loads per v_wmma).
// mode 0: out = bf16( softplus(x) - log2 )   mode 1: out = f32(x)
__global__ void __launch_bounds__(256)
gemm_bf16_kernel(const unsigned short* __restrict__ A,
                 const unsigned short* __restrict__ Bt,
                 const float* __restrict__ bias,
                 void* __restrict__ out, int M, int mode) {
  const int lane = threadIdx.x & 31;
  const int wave = threadIdx.x >> 5;
  const int rowBase = blockIdx.x * 128 + wave * 16;
  const int colBase = blockIdx.y * 64;
  if (rowBase >= M) return;                       // wave-uniform

  const int arow = rowBase + (lane & 15);
  const int brow0 = colBase + (lane & 15);
  __builtin_prefetch(A + (size_t)arow * HDIM, 0, 1);    // global_prefetch path

  v8f c0 = {}, c1 = {}, c2 = {}, c3 = {};
#pragma unroll
  for (int k0 = 0; k0 < HDIM; k0 += 32) {
    v16bf a  = load_a_frag(A, arow, k0, lane);
    v16bf b0 = load_b_frag(Bt, brow0,      k0, lane);
    v16bf b1 = load_b_frag(Bt, brow0 + 16, k0, lane);
    v16bf b2 = load_b_frag(Bt, brow0 + 32, k0, lane);
    v16bf b3 = load_b_frag(Bt, brow0 + 48, k0, lane);
    c0 = __builtin_amdgcn_wmma_f32_16x16x32_bf16(false, a, false, b0, (short)0, c0, false, false);
    c1 = __builtin_amdgcn_wmma_f32_16x16x32_bf16(false, a, false, b1, (short)0, c1, false, false);
    c2 = __builtin_amdgcn_wmma_f32_16x16x32_bf16(false, a, false, b2, (short)0, c2, false, false);
    c3 = __builtin_amdgcn_wmma_f32_16x16x32_bf16(false, a, false, b3, (short)0, c3, false, false);
  }

  const int mBase = rowBase + ((lane >> 4) << 3);
  v8f accs[4] = {c0, c1, c2, c3};
#pragma unroll
  for (int t = 0; t < 4; ++t) {
    const int nCol = colBase + t * 16 + (lane & 15);
    const float bv = bias ? bias[nCol] : 0.0f;
    if (mode == 0) {
      unsigned short* O = (unsigned short*)out;
#pragma unroll
      for (int r = 0; r < 8; ++r) {
        float x = accs[t][r] + bv;
        float sp = (x > 20.0f) ? x : log1pf(expf(x));
        O[(size_t)(mBase + r) * HDIM + nCol] = f2bf(sp - LOG2_F);
      }
    } else {
      float* O = (float*)out;
#pragma unroll
      for (int r = 0; r < 8; ++r)
        O[(size_t)(mBase + r) * HDIM + nCol] = accs[t][r] + bv;
    }
  }
}

// ---------------------- row normalization into z-sets ----------------------
// one block per sorted position p: writes z_s and z_v[0..2] rows (bf16)
__global__ void __launch_bounds__(256)
normalize_rows_kernel(const float* __restrict__ SI, const float* __restrict__ VI,
                      const int* __restrict__ order,
                      unsigned short* __restrict__ Z, int zStride) {
  const int p = blockIdx.x;
  const int h = threadIdx.x;
  const int i = order[p];
  float s  = SI[(size_t)i * HDIM + h];
  float v0 = VI[(size_t)i * 768 + h];
  float v1 = VI[(size_t)i * 768 + 256 + h];
  float v2 = VI[(size_t)i * 768 + 512 + h];
  __shared__ float red[4][256];
  red[0][h] = s * s; red[1][h] = v0 * v0; red[2][h] = v1 * v1; red[3][h] = v2 * v2;
  __syncthreads();
  for (int off = 128; off > 0; off >>= 1) {
    if (h < off) {
      red[0][h] += red[0][h + off]; red[1][h] += red[1][h + off];
      red[2][h] += red[2][h + off]; red[3][h] += red[3][h + off];
    }
    __syncthreads();
  }
  float r0 = 1.0f / fmaxf(sqrtf(red[0][0]), 1e-12f);
  float r1 = 1.0f / fmaxf(sqrtf(red[1][0]), 1e-12f);
  float r2 = 1.0f / fmaxf(sqrtf(red[2][0]), 1e-12f);
  float r3 = 1.0f / fmaxf(sqrtf(red[3][0]), 1e-12f);
  size_t rowOff = (size_t)p * HDIM + h;
  Z[rowOff]                       = f2bf(s * r0);
  Z[(size_t)1 * zStride + rowOff] = f2bf(v0 * r1);
  Z[(size_t)2 * zStride + rowOff] = f2bf(v1 * r2);
  Z[(size_t)3 * zStride + rowOff] = f2bf(v2 * r3);
}

// ---------------------- per-fragment variance stats ------------------------
__global__ void __launch_bounds__(256)
frag_stats_kernel(const float* __restrict__ SI, const float* __restrict__ VI,
                  const int* __restrict__ order, const int* __restrict__ starts,
                  const int* __restrict__ counts,
                  float* __restrict__ s_var, float* __restrict__ v_var) {
  const int f = blockIdx.x;
  const int h = threadIdx.x;
  const int st = starts[f], n = counts[f];
  float ssum = 0, ssq = 0, vs0 = 0, vs1 = 0, vs2 = 0, vsq = 0;
  for (int t = 0; t < n; ++t) {                 // fixed order -> deterministic
    int i = order[st + t];
    float s = SI[(size_t)i * HDIM + h];
    ssum += s; ssq += s * s;
    float a = VI[(size_t)i * 768 + h];
    float b = VI[(size_t)i * 768 + 256 + h];
    float c = VI[(size_t)i * 768 + 512 + h];
    vs0 += a; vs1 += b; vs2 += c;
    vsq += a * a + b * b + c * c;
  }
  __shared__ float r1[256], r2[256], r3[256], r4[256];
  r1[h] = ssq; r2[h] = ssum * ssum;
  r3[h] = vsq; r4[h] = vs0 * vs0 + vs1 * vs1 + vs2 * vs2;
  __syncthreads();
  for (int off = 128; off > 0; off >>= 1) {
    if (h < off) {
      r1[h] += r1[h + off]; r2[h] += r2[h + off];
      r3[h] += r3[h + off]; r4[h] += r4[h + off];
    }
    __syncthreads();
  }
  if (h == 0) {
    if (n > 0) {
      float nf = (float)n;
      s_var[f] = (r1[0] - r2[0] / nf) / nf;
      v_var[f] = (r3[0] - r4[0] / nf) / nf;
    } else { s_var[f] = 0.0f; v_var[f] = 0.0f; }
  }
}

// ---------------- per-fragment Gram pair error (WMMA) ----------------------
// grid (NFRAG, 4 z-sets); each wave walks upper-tri 16x16 tile pairs
__global__ void __launch_bounds__(256)
gram_pairs_kernel(const unsigned short* __restrict__ Zbase, int zStride,
                  const int* __restrict__ starts, const int* __restrict__ counts,
                  float* __restrict__ pair_err) {
  const int f = blockIdx.x;
  const int s = blockIdx.y;
  const unsigned short* Z = Zbase + (size_t)s * zStride;
  const int st = starts[f], n = counts[f];
  const int lane = threadIdx.x & 31;
  const int wave = threadIdx.x >> 5;
  __shared__ float part[8];
  float acc = 0.0f;
  if (n >= 2) {
    const int nt = (n + 15) >> 4;
    const int npair = (nt * (nt + 1)) >> 1;
    for (int t = wave; t < npair; t += 8) {       // wave-uniform loop
      int tt = t, ti = 0, span = nt;
      while (tt >= span) { tt -= span; ti++; span--; }
      int tj = ti + tt;                           // ti <= tj
      int arow = st + ti * 16 + (lane & 15);
      int brow = st + tj * 16 + (lane & 15);
      v8f c = {};
#pragma unroll
      for (int k0 = 0; k0 < HDIM; k0 += 32) {
        v16bf a = load_a_frag(Z, arow, k0, lane);
        v16bf b = load_b_frag(Z, brow, k0, lane);
        c = __builtin_amdgcn_wmma_f32_16x16x32_bf16(false, a, false, b, (short)0, c,
                                                    false, false);
      }
      int jloc = tj * 16 + (lane & 15);
      int ibase = ti * 16 + ((lane >> 4) << 3);
#pragma unroll
      for (int r = 0; r < 8; ++r) {
        int iloc = ibase + r;
        if (iloc < jloc && jloc < n) {
          float d = c[r] - C_CONS;
          acc += d * d;
        }
      }
    }
  }
  for (int off = 16; off > 0; off >>= 1) acc += __shfl_down(acc, off, 32);
  if (lane == 0) part[wave] = acc;
  __syncthreads();
  if (threadIdx.x == 0) {
    float tot = 0.0f;
    for (int w = 0; w < 8; ++w) tot += part[w];
    pair_err[s * NFRAG + f] = tot;
  }
}

// ------------------------------- finalize ----------------------------------
__global__ void __launch_bounds__(256)
finalize_kernel(const int* __restrict__ counts, const float* __restrict__ s_var,
                const float* __restrict__ v_var, const float* __restrict__ pair_err,
                float* __restrict__ out) {
  const int f = threadIdx.x;
  float n = (float)counts[f];
  float pc = 0.5f * n * (n - 1.0f);
  float pcs = fmaxf(pc, 1.0f);
  float sl = s_var[f] + pair_err[0 * NFRAG + f] / pcs;
  float de = pair_err[1 * NFRAG + f] + pair_err[2 * NFRAG + f] + pair_err[3 * NFRAG + f];
  float vl = v_var[f] + de / (3.0f * pcs);
  float fl = 0.5f * sl + 0.5f * vl;               // (1-VW)=VW=0.5
  bool valid = pc > 0.0f;
  __shared__ float rs[256], rc[256];
  rs[f] = valid ? fl : 0.0f;
  rc[f] = valid ? 1.0f : 0.0f;
  __syncthreads();
  for (int off = 128; off > 0; off >>= 1) {
    if (f < off) { rs[f] += rs[f + off]; rc[f] += rc[f + off]; }
    __syncthreads();
  }
  if (f == 0) out[0] = (rc[0] > 0.0f) ? rs[0] / fmaxf(rc[0], 1.0f) : 0.0f;
}

// ------------------------------ launcher -----------------------------------
extern "C" void kernel_launch(void* const* d_in, const int* in_sizes, int n_in,
                              void* d_out, int out_size, void* d_ws, size_t ws_size,
                              hipStream_t stream) {
  const float* ss = (const float*)d_in[0];
  const float* sl = (const float*)d_in[1];
  const float* vs = (const float*)d_in[2];
  const float* vl = (const float*)d_in[3];
  const float* W1 = (const float*)d_in[4];
  const float* b1 = (const float*)d_in[5];
  const float* W2 = (const float*)d_in[6];
  const float* b2 = (const float*)d_in[7];
  const float* V1 = (const float*)d_in[8];
  const float* V2 = (const float*)d_in[9];
  const int* frag = (const int*)d_in[10];

  const int n = in_sizes[0] / HDIM;               // 8192
  const int n3 = 3 * n;
  const int npad = n + 64;
  const int zStride = npad * HDIM;                // elements per z-set

  char* ws = (char*)d_ws;
  size_t off = 0;
  auto take = [&](size_t bytes) -> char* {
    char* p = ws + off;
    off = (off + bytes + 255) & ~(size_t)255;
    return p;
  };
  unsigned short* wt1 = (unsigned short*)take((size_t)HDIM * HDIM * 2);
  unsigned short* wt2 = (unsigned short*)take((size_t)HDIM * HDIM * 2);
  unsigned short* vt1 = (unsigned short*)take((size_t)HDIM * HDIM * 2);
  unsigned short* vt2 = (unsigned short*)take((size_t)HDIM * HDIM * 2);
  unsigned short* Xs  = (unsigned short*)take((size_t)n  * HDIM * 2);
  unsigned short* Xv  = (unsigned short*)take((size_t)n3 * HDIM * 2);
  unsigned short* H1s = (unsigned short*)take((size_t)n  * HDIM * 2);
  unsigned short* H1v = (unsigned short*)take((size_t)n3 * HDIM * 2);
  float* SI = (float*)take((size_t)n  * HDIM * 4);
  float* VI = (float*)take((size_t)n3 * HDIM * 4);
  unsigned short* Z = (unsigned short*)take((size_t)4 * zStride * 2);
  int* counts = (int*)take(NFRAG * 4);
  int* starts = (int*)take(NFRAG * 4);
  int* order  = (int*)take((size_t)n * 4);
  float* s_var = (float*)take(NFRAG * 4);
  float* v_var = (float*)take(NFRAG * 4);
  float* pair_err = (float*)take(4 * NFRAG * 4);

  // prep
  transpose_bf16_kernel<<<256, 256, 0, stream>>>(W1, wt1);
  transpose_bf16_kernel<<<256, 256, 0, stream>>>(W2, wt2);
  transpose_bf16_kernel<<<256, 256, 0, stream>>>(V1, vt1);
  transpose_bf16_kernel<<<256, 256, 0, stream>>>(V2, vt2);
  mix_bf16_kernel<<<2048, 256, 0, stream>>>(ss, sl, Xs, n * HDIM);
  mix_bf16_kernel<<<2048, 256, 0, stream>>>(vs, vl, Xv, n3 * HDIM);
  zero_tail_kernel<<<256, 256, 0, stream>>>(Z, zStride, n);

  // deterministic counting sort by fragment
  hist_scan_kernel<<<1, 256, 0, stream>>>(frag, counts, starts, n);
  rank_scatter_kernel<<<(n + 255) / 256, 256, 0, stream>>>(frag, starts, order, n);

  // MLPs via bf16 WMMA (16x64 strip per wave, 4 col-tile groups)
  gemm_bf16_kernel<<<dim3(n / 128, 4), 256, 0, stream>>>(Xs,  wt1, b1, H1s, n,  0);
  gemm_bf16_kernel<<<dim3(n / 128, 4), 256, 0, stream>>>(H1s, wt2, b2, SI,  n,  1);
  gemm_bf16_kernel<<<dim3(n3 / 128, 4), 256, 0, stream>>>(Xv,  vt1, nullptr, H1v, n3, 0);
  gemm_bf16_kernel<<<dim3(n3 / 128, 4), 256, 0, stream>>>(H1v, vt2, nullptr, VI,  n3, 1);

  // normalized z rows in sorted order
  normalize_rows_kernel<<<n, 256, 0, stream>>>(SI, VI, order, Z, zStride);

  // per-fragment stats + WMMA Gram pair errors
  frag_stats_kernel<<<NFRAG, 256, 0, stream>>>(SI, VI, order, starts, counts, s_var, v_var);
  gram_pairs_kernel<<<dim3(NFRAG, 4), 256, 0, stream>>>(Z, zStride, starts, counts, pair_err);

  finalize_kernel<<<1, 256, 0, stream>>>(counts, s_var, v_var, pair_err, (float*)d_out);
}
